// T5EncoderLayer_77927886619151
// MI455X (gfx1250) — matrix operand order repaired and checked by reading, not compiled, not run on hardware
//
#include <hip/hip_runtime.h>
#include <hip/hip_bf16.h>
#include <math.h>

// ---- problem constants (match reference) ----
#define Bb    8
#define Ls    1024
#define Dd    1024
#define Hh    16
#define DHh   64
#define HDH   (Hh*DHh)     // 1024
#define DFFs  2816
#define NBUCK 32
#define MAXDIST 128

typedef __attribute__((ext_vector_type(16))) __bf16 v16bf;
typedef __attribute__((ext_vector_type(8)))  __bf16 v8bf;
typedef __attribute__((ext_vector_type(8)))  float  v8f;

// ---------- helpers ----------
static __device__ __forceinline__ __bf16 f2bf(float f) {
  unsigned int u = __builtin_bit_cast(unsigned int, f);
  u += 0x7FFFu + ((u >> 16) & 1u);          // round-to-nearest-even
  unsigned short s = (unsigned short)(u >> 16);
  return __builtin_bit_cast(__bf16, s);
}

static __device__ __forceinline__ v8f vzero8() {
  v8f z;
#pragma unroll
  for (int i = 0; i < 8; ++i) z[i] = 0.0f;
  return z;
}

static __device__ __forceinline__ v16bf join16(v8bf lo, v8bf hi) {
  return __builtin_shufflevector(lo, hi, 0,1,2,3,4,5,6,7,8,9,10,11,12,13,14,15);
}

static __device__ __forceinline__ v8f wmma_bf16(v16bf a, v16bf b, v8f c) {
  // D = A(16x32 bf16) * B(32x16 bf16) + C(16x16 f32)
  return __builtin_amdgcn_wmma_f32_16x16x32_bf16(
      false, a, false, b, (short)0, c, false, false);
}

static __device__ __forceinline__ float gelu_tanh(float x) {
  const float k0 = 0.7978845608028654f;     // sqrt(2/pi)
  float x3 = x * x * x;
  float t = tanhf(k0 * (x + 0.044715f * x3));
  return 0.5f * x * (1.0f + t);
}

static __device__ __forceinline__ int rp_bucket(int rel) {
  const int nb = NBUCK / 2;                 // 16
  int b  = (rel > 0) ? nb : 0;
  int rp = (rel < 0) ? -rel : rel;
  const int me = nb / 2;                    // 8
  if (rp < me) return b + rp;
  float lf = __logf((float)rp / (float)me) / __logf((float)MAXDIST / (float)me);
  int large = me + (int)(lf * (float)(nb - me));
  large = min(large, nb - 1);
  return b + large;
}

// ---------- weight transpose + bf16 convert: W[K][N] f32 -> WT[N][K] bf16 ----
__global__ __launch_bounds__(256)
void transpose_w_kernel(const float* __restrict__ W, __bf16* __restrict__ WT,
                        int K, int N) {
  __shared__ float tile[32][33];
  const int kb = blockIdx.x * 32, nb = blockIdx.y * 32;
  const int tx = threadIdx.x & 31, ty = threadIdx.x >> 5;   // 8 rows per pass
#pragma unroll
  for (int r = ty; r < 32; r += 8)
    tile[r][tx] = W[(size_t)(kb + r) * N + nb + tx];
  __syncthreads();
#pragma unroll
  for (int r = ty; r < 32; r += 8)
    WT[(size_t)(nb + r) * K + kb + tx] = f2bf(tile[tx][r]);
}

// ---------- expand relative-position bias: tab[h][rel+1023] ----------
__global__ __launch_bounds__(256)
void rpb_expand_kernel(const float* __restrict__ rpb, float* __restrict__ tab) {
  const int i = blockIdx.x * 256 + threadIdx.x;   // over Hh*2048
  if (i >= Hh * 2048) return;
  const int h = i >> 11;
  const int rel = (i & 2047) - 1023;
  tab[i] = rpb[rp_bucket(rel) * Hh + h];
}

// ---------- RMSNorm (T5): y_bf16 = x * rsqrt(mean(x^2)+eps) * w ----------
__global__ __launch_bounds__(256)
void rmsnorm_kernel(const float* __restrict__ X, const float* __restrict__ W,
                    __bf16* __restrict__ Y) {
  const int row = blockIdx.x;
  const float* x = X + (size_t)row * Dd;
  float ss = 0.f;
  for (int i = threadIdx.x; i < Dd; i += 256) { float v = x[i]; ss += v * v; }
#pragma unroll
  for (int off = 16; off > 0; off >>= 1) ss += __shfl_xor(ss, off, 32);
  __shared__ float red[8];
  const int wave = threadIdx.x >> 5, lane = threadIdx.x & 31;
  if (lane == 0) red[wave] = ss;
  __syncthreads();
  if (wave == 0) {
    float v = (lane < 8) ? red[lane] : 0.f;
#pragma unroll
    for (int off = 4; off > 0; off >>= 1) v += __shfl_xor(v, off, 32);
    if (lane == 0) red[0] = v;
  }
  __syncthreads();
  const float inv = rsqrtf(red[0] / (float)Dd + 1e-6f);
  __bf16* y = Y + (size_t)row * Dd;
  for (int i = threadIdx.x; i < Dd; i += 256)
    y[i] = f2bf(x[i] * inv * W[i]);
}

// ---------- generic WMMA GEMM: C[M,N] = A_bf16[M,K] * BT_bf16[N][K] (+Res) --
// one wave -> 32(M) x 64(N) tile (2 A-frags share each B-frag, 8 accums);
// requires M%32==0, N%64==0, K%32==0
// OUTMODE: 0 = f32 row-major, 1 = bf16 row-major,
//          2 = bf16 "V-transposed" [B,H,DH,L]  (row=(b,l), col=(h,dh))
template<bool RES, int OUTMODE>
__global__ __launch_bounds__(32)
void gemm_bt_kernel(const __bf16* __restrict__ A, const __bf16* __restrict__ BT,
                    const float* __restrict__ Res, void* __restrict__ Cout,
                    int M, int N, int K) {
  const int lane = threadIdx.x & 31;
  const int n16  = lane & 15;
  const int half = lane >> 4;
  const int row0 = blockIdx.y * 32;
  const int col0 = blockIdx.x * 64;

  v8f acc[2][4];
#pragma unroll
  for (int mt = 0; mt < 2; ++mt)
#pragma unroll
    for (int i = 0; i < 4; ++i) acc[mt][i] = vzero8();

  const __bf16* Arow0 = A + (size_t)(row0 + n16) * K;       // A-frag m = lane&15
  const __bf16* Arow1 = A + (size_t)(row0 + 16 + n16) * K;
  const __bf16* Bc0   = BT + (size_t)(col0 + n16) * K;      // B-frag n = lane&15
  const size_t  bstep = (size_t)16 * K;                     // next 16-col tile

  for (int k0 = 0; k0 < K; k0 += 32) {
    v16bf a0 = join16(*(const v8bf*)(Arow0 + k0 + half * 16),
                      *(const v8bf*)(Arow0 + k0 + half * 16 + 8));
    v16bf a1 = join16(*(const v8bf*)(Arow1 + k0 + half * 16),
                      *(const v8bf*)(Arow1 + k0 + half * 16 + 8));
#pragma unroll
    for (int nt = 0; nt < 4; ++nt) {
      v16bf bb = *(const v16bf*)(Bc0 + (size_t)nt * bstep + k0 + half * 16);
      acc[0][nt] = wmma_bf16(a0, bb, acc[0][nt]);
      acc[1][nt] = wmma_bf16(a1, bb, acc[1][nt]);
    }
  }
#pragma unroll
  for (int mt = 0; mt < 2; ++mt) {
#pragma unroll
    for (int nt = 0; nt < 4; ++nt) {
#pragma unroll
      for (int r = 0; r < 8; ++r) {
        const int row = row0 + mt * 16 + half * 8 + r;
        const int col = col0 + nt * 16 + n16;
        float v = acc[mt][nt][r];
        if (RES) v += Res[(size_t)row * N + col];
        if (OUTMODE == 0) {
          ((float*)Cout)[(size_t)row * N + col] = v;
        } else if (OUTMODE == 1) {
          ((__bf16*)Cout)[(size_t)row * N + col] = f2bf(v);
        } else {
          const int b = row >> 10, l = row & (Ls - 1);
          const int h = col >> 6,  d = col & (DHh - 1);
          ((__bf16*)Cout)[(((size_t)b * Hh + h) * DHh + d) * Ls + l] = f2bf(v);
        }
      }
    }
  }
}

// ---------- fused gated FFN GEMM: h2 = gelu(A*W0) * (A*W1), bf16 out -------
__global__ __launch_bounds__(32)
void ffn_gate_kernel(const __bf16* __restrict__ A, const __bf16* __restrict__ BT0,
                     const __bf16* __restrict__ BT1, __bf16* __restrict__ Hout,
                     int M, int N, int K) {
  const int lane = threadIdx.x & 31;
  const int n16  = lane & 15;
  const int half = lane >> 4;
  const int row0 = blockIdx.y * 16;
  const int col0 = blockIdx.x * 64;

  v8f acc0[4], acc1[4];
#pragma unroll
  for (int i = 0; i < 4; ++i) { acc0[i] = vzero8(); acc1[i] = vzero8(); }

  const __bf16* Arow = A + (size_t)(row0 + n16) * K;
  const __bf16* B0c  = BT0 + (size_t)(col0 + n16) * K;
  const __bf16* B1c  = BT1 + (size_t)(col0 + n16) * K;
  const size_t  bstep = (size_t)16 * K;

  for (int k0 = 0; k0 < K; k0 += 32) {
    v16bf a = join16(*(const v8bf*)(Arow + k0 + half * 16),
                     *(const v8bf*)(Arow + k0 + half * 16 + 8));
#pragma unroll
    for (int nt = 0; nt < 4; ++nt) {
      v16bf b0 = *(const v16bf*)(B0c + (size_t)nt * bstep + k0 + half * 16);
      v16bf b1 = *(const v16bf*)(B1c + (size_t)nt * bstep + k0 + half * 16);
      acc0[nt] = wmma_bf16(a, b0, acc0[nt]);
      acc1[nt] = wmma_bf16(a, b1, acc1[nt]);
    }
  }
#pragma unroll
  for (int nt = 0; nt < 4; ++nt) {
#pragma unroll
    for (int r = 0; r < 8; ++r) {
      const int row = row0 + half * 8 + r;
      const int col = col0 + nt * 16 + n16;
      const float g = gelu_tanh(acc0[nt][r]);
      Hout[(size_t)row * N + col] = f2bf(g * acc1[nt][r]);
    }
  }
}

// ---------- flash attention with T5 relative-position bias ----------
// one wave per (b, h, 16-query tile).
// Q,K bf16 [B,L,H,DH]; V bf16 transposed [B,H,DH,L]; bias table [H][2048].
__global__ __launch_bounds__(32)
void attn_kernel(const __bf16* __restrict__ Q, const __bf16* __restrict__ Kb,
                 const __bf16* __restrict__ VT, const float* __restrict__ btab,
                 const int* __restrict__ pad, __bf16* __restrict__ Ctx) {
  const int tile = blockIdx.x;
  const int qt = tile & (Ls / 16 - 1);
  const int h  = (tile / (Ls / 16)) & (Hh - 1);
  const int b  = tile / (Ls / 16) / Hh;

  const int lane = threadIdx.x & 31;
  const int n16  = lane & 15;
  const int half = lane >> 4;
  const int m    = n16;                     // A-fragment row role
  const int q0   = qt * 16;
  const size_t baseBH = (size_t)b * Ls * HDH + (size_t)h * DHh;
  const float* brow = btab + h * 2048 + 1023;

  // Q as two 16x32 A-fragments covering DH=64
  const __bf16* qrow = Q + baseBH + (size_t)(q0 + m) * HDH;
  v16bf aq[2];
#pragma unroll
  for (int kb = 0; kb < 2; ++kb) {
    v8bf lo = *(const v8bf*)(qrow + kb * 32 + half * 16);
    v8bf hi = *(const v8bf*)(qrow + kb * 32 + half * 16 + 8);
    aq[kb] = join16(lo, hi);
  }

  v8f ctx[4];
#pragma unroll
  for (int i = 0; i < 4; ++i) ctx[i] = vzero8();
  float rmax[8], rsum[8];
#pragma unroll
  for (int r = 0; r < 8; ++r) { rmax[r] = -3.0e38f; rsum[r] = 0.f; }

  __shared__ __bf16 pbuf[16][32];

  for (int kc = 0; kc < Ls; kc += 32) {
    // ---- scores S = Q * K^T for 32 keys (two 16-key column tiles) ----
    v8f S0 = vzero8(), S1 = vzero8();
    const __bf16* krow0 = Kb + baseBH + (size_t)(kc + n16) * HDH;
    const __bf16* krow1 = Kb + baseBH + (size_t)(kc + 16 + n16) * HDH;
#pragma unroll
    for (int kb = 0; kb < 2; ++kb) {
      v16bf bk0 = *(const v16bf*)(krow0 + kb * 32 + half * 16);
      v16bf bk1 = *(const v16bf*)(krow1 + kb * 32 + half * 16);
      S0 = wmma_bf16(aq[kb], bk0, S0);
      S1 = wmma_bf16(aq[kb], bk1, S1);
    }

    // ---- bias + online softmax ----
    const float pb0 = (pad[b * Ls + kc + n16]      > 0) ? 0.f : -1e9f;
    const float pb1 = (pad[b * Ls + kc + 16 + n16] > 0) ? 0.f : -1e9f;
#pragma unroll
    for (int r = 0; r < 8; ++r) {
      const int qpos = q0 + half * 8 + r;
      float s0 = S0[r] + brow[kc + n16      - qpos] + pb0;
      float s1 = S1[r] + brow[kc + 16 + n16 - qpos] + pb1;
      float mx = fmaxf(s0, s1);
#pragma unroll
      for (int off = 1; off < 16; off <<= 1) mx = fmaxf(mx, __shfl_xor(mx, off, 32));
      const float nm = fmaxf(rmax[r], mx);
      const float sc = __expf(rmax[r] - nm);
      const float p0 = __expf(s0 - nm);
      const float p1 = __expf(s1 - nm);
      float ps = p0 + p1;
#pragma unroll
      for (int off = 1; off < 16; off <<= 1) ps += __shfl_xor(ps, off, 32);
      rsum[r] = rsum[r] * sc + ps;
      rmax[r] = nm;
#pragma unroll
      for (int t = 0; t < 4; ++t) ctx[t][r] *= sc;
      pbuf[half * 8 + r][n16]      = f2bf(p0);
      pbuf[half * 8 + r][16 + n16] = f2bf(p1);
    }
    __syncthreads();                         // single wave: ordering point

    // ---- P (16x32) as A-fragment from LDS ----
    v8bf plo = *(const v8bf*)(&pbuf[m][half * 16]);
    v8bf phi = *(const v8bf*)(&pbuf[m][half * 16 + 8]);
    v16bf aP = join16(plo, phi);
    __syncthreads();

    // ---- ctx += P * V (32 keys x 64 dims); V is [B,H,DH,L] ----
    const __bf16* vbase = VT + (((size_t)b * Hh + h) * DHh) * Ls;
#pragma unroll
    for (int nt = 0; nt < 4; ++nt) {
      v16bf bv = *(const v16bf*)(vbase + (size_t)(nt * 16 + n16) * Ls + kc + half * 16);
      ctx[nt] = wmma_bf16(aP, bv, ctx[nt]);
    }
  }

  // ---- normalize and store (row-major [B,L,H*DH] for next GEMM's A) ----
#pragma unroll
  for (int nt = 0; nt < 4; ++nt) {
#pragma unroll
    for (int r = 0; r < 8; ++r) {
      const int qpos = q0 + half * 8 + r;
      Ctx[baseBH + (size_t)qpos * HDH + nt * 16 + n16] = f2bf(ctx[nt][r] / rsum[r]);
    }
  }
}

// ---------- host launch ----------
extern "C" void kernel_launch(void* const* d_in, const int* in_sizes, int n_in,
                              void* d_out, int out_size, void* d_ws, size_t ws_size,
                              hipStream_t stream) {
  const float* hidden_states = (const float*)d_in[0];
  const int*   pad_valid     = (const int*)  d_in[1];
  const float* wq      = (const float*)d_in[2];
  const float* wk      = (const float*)d_in[3];
  const float* wv      = (const float*)d_in[4];
  const float* wo_attn = (const float*)d_in[5];
  const float* rpb     = (const float*)d_in[6];
  const float* ln1     = (const float*)d_in[7];
  const float* wi0     = (const float*)d_in[8];
  const float* wi1     = (const float*)d_in[9];
  const float* wo_ff   = (const float*)d_in[10];
  const float* ln2     = (const float*)d_in[11];

  const size_t ML = (size_t)Bb * Ls;        // 8192 rows
  char* ws = (char*)d_ws;
  size_t off = 0;
  auto alloc = [&](size_t bytes) -> char* {
    char* p = ws + off;
    off += (bytes + 255) & ~(size_t)255;
    return p;
  };
  __bf16* xn    = (__bf16*)alloc(ML * Dd   * 2);  // rmsnorm out (reused as y)
  __bf16* qb    = (__bf16*)alloc(ML * HDH  * 2);
  __bf16* kbuf  = (__bf16*)alloc(ML * HDH  * 2);
  __bf16* vT    = (__bf16*)alloc(ML * HDH  * 2);  // [B,H,DH,L]
  __bf16* ctxb  = (__bf16*)alloc(ML * HDH  * 2);
  float*  hid   = (float*) alloc(ML * Dd   * 4);
  __bf16* h2    = (__bf16*)alloc(ML * DFFs * 2);
  __bf16* wqT   = (__bf16*)alloc((size_t)HDH  * Dd   * 2);  // [N][K]
  __bf16* wkT   = (__bf16*)alloc((size_t)HDH  * Dd   * 2);
  __bf16* wvT   = (__bf16*)alloc((size_t)HDH  * Dd   * 2);
  __bf16* woaT  = (__bf16*)alloc((size_t)Dd   * HDH  * 2);
  __bf16* wi0T  = (__bf16*)alloc((size_t)DFFs * Dd   * 2);
  __bf16* wi1T  = (__bf16*)alloc((size_t)DFFs * Dd   * 2);
  __bf16* woffT = (__bf16*)alloc((size_t)Dd   * DFFs * 2);
  float*  btab  = (float*) alloc((size_t)Hh * 2048 * 4);
  (void)ws_size; (void)in_sizes; (void)n_in; (void)out_size;

  // ---- one-time prep: weight transposes (f32 -> bf16 [N][K]) + bias table --
  transpose_w_kernel<<<dim3(Dd/32,  HDH/32),  256, 0, stream>>>(wq,      wqT,   Dd,   HDH);
  transpose_w_kernel<<<dim3(Dd/32,  HDH/32),  256, 0, stream>>>(wk,      wkT,   Dd,   HDH);
  transpose_w_kernel<<<dim3(Dd/32,  HDH/32),  256, 0, stream>>>(wv,      wvT,   Dd,   HDH);
  transpose_w_kernel<<<dim3(HDH/32, Dd/32),   256, 0, stream>>>(wo_attn, woaT,  HDH,  Dd);
  transpose_w_kernel<<<dim3(Dd/32,  DFFs/32), 256, 0, stream>>>(wi0,     wi0T,  Dd,   DFFs);
  transpose_w_kernel<<<dim3(Dd/32,  DFFs/32), 256, 0, stream>>>(wi1,     wi1T,  Dd,   DFFs);
  transpose_w_kernel<<<dim3(DFFs/32, Dd/32),  256, 0, stream>>>(wo_ff,   woffT, DFFs, Dd);
  rpb_expand_kernel<<<(Hh * 2048) / 256, 256, 0, stream>>>(rpb, btab);

  const dim3 blkGemm(32);
  const dim3 grdProj(HDH / 64, (unsigned)(ML / 32));   // N=1024, 32-row tiles
  const dim3 grdFFN (DFFs / 64, (unsigned)(ML / 16));  // N=2816

  // 1) x = rmsnorm(hidden_states, ln1) -> bf16
  rmsnorm_kernel<<<(unsigned)ML, 256, 0, stream>>>(hidden_states, ln1, xn);

  // 2) q/k/v projections (bf16 out; v stored [B,H,DH,L])
  gemm_bt_kernel<false, 1><<<grdProj, blkGemm, 0, stream>>>(xn, wqT, nullptr, qb,   (int)ML, HDH, Dd);
  gemm_bt_kernel<false, 1><<<grdProj, blkGemm, 0, stream>>>(xn, wkT, nullptr, kbuf, (int)ML, HDH, Dd);
  gemm_bt_kernel<false, 2><<<grdProj, blkGemm, 0, stream>>>(xn, wvT, nullptr, vT,   (int)ML, HDH, Dd);

  // 3) attention (flash, with T5 relative position bias + pad mask)
  attn_kernel<<<(unsigned)(Bb * Hh * (Ls / 16)), 32, 0, stream>>>(qb, kbuf, vT, btab, pad_valid, ctxb);

  // 4) hidden = hidden_states + ctx @ wo_attn (f32 out)
  gemm_bt_kernel<true, 0><<<grdProj, blkGemm, 0, stream>>>(ctxb, woaT, hidden_states, hid, (int)ML, Dd, HDH);

  // 5) y = rmsnorm(hidden, ln2) -> bf16 (reuse xn buffer)
  rmsnorm_kernel<<<(unsigned)ML, 256, 0, stream>>>(hid, ln2, xn);

  // 6) h2 = gelu_tanh(y@wi0) * (y@wi1) -> bf16
  ffn_gate_kernel<<<grdFFN, blkGemm, 0, stream>>>(xn, wi0T, wi1T, h2, (int)ML, DFFs, Dd);

  // 7) out = hidden + h2 @ wo_ff (f32 out)
  gemm_bt_kernel<true, 0><<<grdProj, blkGemm, 0, stream>>>(h2, woffT, hid, (float*)d_out, (int)ML, Dd, DFFs);
}
